// Attention_42253888258100
// MI455X (gfx1250) — matrix operand order repaired and checked by reading, not compiled, not run on hardware
//
#include <hip/hip_runtime.h>

typedef __attribute__((ext_vector_type(16))) _Float16 v16h;
typedef __attribute__((ext_vector_type(8)))  _Float16 h8v;
typedef __attribute__((ext_vector_type(8)))  float    v8f;

#define SDIM 2048
#define DDIM 64
#define QT   16
#define WAVES 4
#define NTHREADS (WAVES * 32)

__global__ __launch_bounds__(NTHREADS) void attn_fwd_f16wmma(
    const float* __restrict__ Q, const float* __restrict__ K,
    const float* __restrict__ V, const int* __restrict__ mask,
    float* __restrict__ Out, float* __restrict__ P)
{
    // Full 16-row score block as f16: 16*2048*2 = 64 KB of LDS.
    __shared__ _Float16 sP[QT * SDIM];

    const int lane = threadIdx.x & 31;
    const int wave = threadIdx.x >> 5;
    const int m16  = lane & 15;   // row (A/C) or col (B/C) within tile
    const int h2   = lane >> 4;   // half-wave select

    const int qt = blockIdx.x;    // 0..127 query tiles
    const int bh = blockIdx.y;    // 0..31 (b*H + h)
    const int b  = bh >> 4;       // H = 16
    const int qb = qt * QT;

    const float* Qb = Q + (size_t)bh * SDIM * DDIM;
    const float* Kb = K + (size_t)bh * SDIM * DDIM;
    const float* Vb = V + (size_t)bh * SDIM * DDIM;
    const int*   Mb = mask + (size_t)b * SDIM;
    float* Pb = P   + (size_t)bh * SDIM * SDIM + (size_t)qb * SDIM;
    float* Ob = Out + ((size_t)bh * SDIM + qb) * DDIM;

    const float scale = 0.125f;   // 1/sqrt(64)

    // ---------------- Pass 1: scores = scale * Q K^T (+mask) -> LDS f16 ----
    // A layout (16x32 f16), lane m = l%16:
    //   elements 0..7  <-> K = 8*h2 + (0..7)
    //   elements 8..15 <-> K = 16 + 8*h2 + (0..7)
    // => two contiguous 8-float runs per half; reused across all 128 k-tiles.
    v16h aq[2];
    #pragma unroll
    for (int s = 0; s < 2; ++s) {
        const float* rp = Qb + (size_t)(qb + m16) * DDIM + s * 32 + 8 * h2;
        #pragma unroll
        for (int half = 0; half < 2; ++half) {
            const float* hp = rp + half * 16;
            #pragma unroll
            for (int e = 0; e < 8; e += 4) {
                float4 f = *(const float4*)(hp + e);
                aq[s][8 * half + e]     = (_Float16)f.x;
                aq[s][8 * half + e + 1] = (_Float16)f.y;
                aq[s][8 * half + e + 2] = (_Float16)f.z;
                aq[s][8 * half + e + 3] = (_Float16)f.w;
            }
        }
    }

    for (int kt = wave; kt < SDIM / 16; kt += WAVES) {
        const int kb = kt * 16;
        v8f c = {};
        #pragma unroll
        for (int s = 0; s < 2; ++s) {
            // B layout (32x16 f16): lane n = l%16; element e -> K = 8*h2 + e + (e & 8)
            // B[kk][n] = K[kb + n][s*32 + kk]  (row of K, contiguous in d)
            v16h bm;
            const float* rp = Kb + (size_t)(kb + m16) * DDIM + s * 32 + 8 * h2;
            #pragma unroll
            for (int e = 0; e < 16; e += 4) {
                float4 f = *(const float4*)(rp + e + (e & 8));
                bm[e]     = (_Float16)f.x;
                bm[e + 1] = (_Float16)f.y;
                bm[e + 2] = (_Float16)f.z;
                bm[e + 3] = (_Float16)f.w;
            }
            c = __builtin_amdgcn_wmma_f32_16x16x32_f16(
                    false, aq[s], false, bm, (short)0, c, false, false);
        }
        // C layout: lane covers col kb+m16, rows r + 8*h2 -> one mask read/lane
        const int mz = Mb[kb + m16];
        #pragma unroll
        for (int r = 0; r < 8; ++r) {
            float sv = c[r] * scale;
            sv = (mz == 0) ? -30000.0f : sv;   // finite in f16; exp -> 0
            sP[(r + 8 * h2) * SDIM + kb + m16] = (_Float16)sv;
        }
    }
    __syncthreads();

    // ---------------- Pass 2: row softmax; write p_attn (f32) + LDS p (f16) --
    #pragma unroll 1
    for (int r = wave * 4; r < wave * 4 + 4; ++r) {
        _Float16* row = &sP[r * SDIM];
        float mx = -1e30f;
        for (int c0 = lane; c0 < SDIM; c0 += 32)
            mx = fmaxf(mx, (float)row[c0]);
        #pragma unroll
        for (int off = 16; off > 0; off >>= 1)
            mx = fmaxf(mx, __shfl_xor(mx, off, 32));

        float sum = 0.f;
        for (int c0 = lane; c0 < SDIM; c0 += 32) {
            float e = __expf((float)row[c0] - mx);
            row[c0] = (_Float16)e;
            sum += e;
        }
        #pragma unroll
        for (int off = 16; off > 0; off >>= 1)
            sum += __shfl_xor(sum, off, 32);
        const float inv = 1.0f / sum;

        float* prow = Pb + (size_t)r * SDIM;
        for (int c0 = lane; c0 < SDIM; c0 += 32) {  // coalesced, non-temporal
            float p = (float)row[c0] * inv;
            __builtin_nontemporal_store(p, prow + c0);  // 537MB stream: keep L2 for K/V
            row[c0]  = (_Float16)p;
        }
    }
    __syncthreads();

    // ---------------- Pass 3: out = P V ; wave owns 16 of the 64 cols -------
    const int nb = wave * 16;
    v8f acc = {};
    for (int kt = 0; kt < SDIM / 32; ++kt) {
        const int k0 = kt * 32;
        // A fragment from LDS f16 probabilities: two ds_load_b128
        v16h ap;
        {
            const _Float16* rp = &sP[m16 * SDIM + k0 + 8 * h2];
            h8v lo = *(const h8v*)(rp);        // elements 0..7
            h8v hi = *(const h8v*)(rp + 16);   // elements 8..15
            #pragma unroll
            for (int e = 0; e < 8; ++e) { ap[e] = lo[e]; ap[8 + e] = hi[e]; }
        }
        // B[kk][n] = V[k0 + kk][nb + n], kk = 8*h2 + e + (e & 8) (stride DDIM over e)
        v16h bv;
        const float* vp = Vb + (size_t)(k0 + 8 * h2) * DDIM + nb + m16;
        if (kt + 1 < SDIM / 32)
            __builtin_prefetch(vp + 32 * DDIM, 0, 0);  // global_prefetch next V tile
        #pragma unroll
        for (int e = 0; e < 16; ++e)
            bv[e] = (_Float16)vp[(size_t)(e + (e & 8)) * DDIM];
        acc = __builtin_amdgcn_wmma_f32_16x16x32_f16(
                  false, ap, false, bv, (short)0, acc, false, false);
    }
    #pragma unroll
    for (int r = 0; r < 8; ++r)
        __builtin_nontemporal_store(acc[r], Ob + (size_t)(r + 8 * h2) * DDIM + nb + m16);
}

extern "C" void kernel_launch(void* const* d_in, const int* in_sizes, int n_in,
                              void* d_out, int out_size, void* d_ws, size_t ws_size,
                              hipStream_t stream) {
    const float* Q    = (const float*)d_in[0];
    const float* K    = (const float*)d_in[1];
    const float* V    = (const float*)d_in[2];
    const int*   mask = (const int*)d_in[3];

    float* Out = (float*)d_out;                                   // [B,H,S,D]
    float* P   = (float*)d_out + (size_t)2 * 16 * 2048 * 64;      // [B,H,S,S]

    dim3 grid(SDIM / QT, 2 * 16);   // 128 q-tiles x (B*H)
    attn_fwd_f16wmma<<<grid, NTHREADS, 0, stream>>>(Q, K, V, mask, Out, P);
}
